// DeformableConv2d_34291018891335
// MI455X (gfx1250) — compile-verified, matrix operand order checked by reading, not run on hardware
//
#include <hip/hip_runtime.h>
#include <hip/hip_bf16.h>
#include <math.h>

#define B_    4
#define CIN_  32
#define CPSF_ 8
#define CXT_  40      // CIN + CPSF
#define COUT_ 64
#define H_    224
#define W_    224
#define HW_   (H_ * W_)
#define KK_   9
#define KDIM_ (CIN_ * KK_)   // 288
#define TILE_N 64
#define RS_   (KDIM_ + 8)    // padded bf16 row stride (296)

typedef __attribute__((ext_vector_type(16))) __bf16 v16bf;
typedef __attribute__((ext_vector_type(8)))  float  v8f;
typedef unsigned int u32x4 __attribute__((ext_vector_type(4)));
typedef int          i32x4 __attribute__((ext_vector_type(4)));
typedef int          i32x8 __attribute__((ext_vector_type(8)));

__device__ __forceinline__ __bf16 f2bf(float f) {
    union { float f; unsigned int u; } in; in.f = f;
    unsigned int u = in.u;
    unsigned int r = (u + 0x7FFFu + ((u >> 16) & 1u)) >> 16;   // RNE
    union { unsigned short s; __bf16 h; } out; out.s = (unsigned short)r;
    return out.h;
}

#if __has_builtin(__builtin_amdgcn_tensor_load_to_lds)
#define HAVE_TDM 1
// Issue one TDM 3D-tile load: tile (tx x ty x tz) elements of f32 from a
// tensor with row stride s0 and plane stride s1 (elements), into LDS at
// byte offset lds_off.  Descriptor bitfields per CDNA5 ISA 8.3-8.6.
__device__ __forceinline__ void tdm_load_tile_3d(const float* gsrc, unsigned lds_off,
                                                 int tx, int ty, int tz,
                                                 unsigned s0, unsigned s1)
{
    unsigned long long ga = (unsigned long long)gsrc;
    u32x4 g0;
    g0[0] = 1u;                                    // count=1 (valid), user mode
    g0[1] = lds_off;                               // lds_addr (bytes)
    g0[2] = (unsigned)(ga & 0xFFFFFFFFu);          // global_addr[31:0]
    g0[3] = (unsigned)((ga >> 32) & 0x01FFFFFFu)   // global_addr[56:32]
          | (2u << 30);                            // type = 2 ("image")
    i32x8 g1;
    g1[0] = (2 << 16);                             // data_size = 4 bytes, no flags
    g1[1] = (int)((W_ & 0xFFFFu) << 16);           // tensor_dim0[15:0]
    g1[2] = (int)(((unsigned)W_ >> 16) | ((W_ & 0xFFFFu) << 16)); // dim0 hi | dim1 lo
    g1[3] = (int)(((unsigned)H_ >> 16) | ((unsigned)tx << 16));   // dim1 hi | tile_dim0
    g1[4] = (int)(((unsigned)ty & 0xFFFFu) | ((unsigned)tz << 16)); // tile_dim1|tile_dim2
    g1[5] = (int)s0;                               // tensor_dim0_stride[31:0]
    g1[6] = (int)((s1 & 0xFFFFu) << 16);           // dim0_stride hi=0 | dim1_stride[15:0]
    g1[7] = (int)(s1 >> 16);                       // tensor_dim1_stride[47:16]
    i32x4 g2;
    g2[0] = tz;                                    // tensor_dim2
    g2[1] = 0; g2[2] = 0; g2[3] = 0;               // no dim3, iterate off
    i32x4 g3; g3[0] = 0; g3[1] = 0; g3[2] = 0; g3[3] = 0;
#if __clang_major__ >= 23
    i32x8 g4; g4[0]=0; g4[1]=0; g4[2]=0; g4[3]=0; g4[4]=0; g4[5]=0; g4[6]=0; g4[7]=0;
    __builtin_amdgcn_tensor_load_to_lds(g0, g1, g2, g3, g4, 0);
#else
    __builtin_amdgcn_tensor_load_to_lds(g0, g1, g2, g3, 0);
#endif
}
#endif

// ---------------------------------------------------------------------------
// Kernel 1: offset/mask 3x3 conv over concat(x, psf), 40 -> 18 + 9 channels.
// 16x16 output tile per block, 18x18x40 halo patch in LDS.
// Interior blocks stage the patch with two Tensor-Data-Mover 3D tile DMAs
// (x: 18x18x32, psf: 18x18x8); boundary blocks use predicated scalar loads.
// ---------------------------------------------------------------------------
__global__ __launch_bounds__(256) void offmask_conv_kernel(
    const float* __restrict__ x, const float* __restrict__ psf,
    const float* __restrict__ ow, const float* __restrict__ ob,
    const float* __restrict__ mw, const float* __restrict__ mb,
    float* __restrict__ offo, float* __restrict__ msko)
{
    __shared__ float patch[CXT_][18][18];   // 51840 B, LDS offset 0 (only object)
    const int b   = blockIdx.z;
    const int ty0 = blockIdx.y * 16, tx0 = blockIdx.x * 16;
    const int tid = threadIdx.y * 16 + threadIdx.x;

#if defined(HAVE_TDM)
    const bool interior = (blockIdx.x > 0) && (blockIdx.x < (W_ / 16 - 1)) &&
                          (blockIdx.y > 0) && (blockIdx.y < (H_ / 16 - 1));
    if (interior) {
        if (tid < 32) {   // one wave issues the DMAs (TDM ignores EXEC)
            const float* xs = x   + ((size_t)b * CIN_)  * HW_ + (ty0 - 1) * W_ + (tx0 - 1);
            const float* ps = psf + ((size_t)b * CPSF_) * HW_ + (ty0 - 1) * W_ + (tx0 - 1);
            tdm_load_tile_3d(xs, 0u,                    18, 18, CIN_,  W_, HW_);
            tdm_load_tile_3d(ps, (unsigned)(CIN_ * 324 * 4), 18, 18, CPSF_, W_, HW_);
#if __has_builtin(__builtin_amdgcn_s_wait_tensorcnt)
            __builtin_amdgcn_s_wait_tensorcnt(0);
#else
            asm volatile("s_wait_tensorcnt 0x0" ::: "memory");
#endif
        }
    } else
#endif
    {
        for (int i = tid; i < CXT_ * 324; i += 256) {
            int c = i / 324, rem = i - c * 324;
            int iy = rem / 18, ix = rem - iy * 18;
            int gy = ty0 - 1 + iy, gx = tx0 - 1 + ix;
            float v = 0.f;
            if (gy >= 0 && gy < H_ && gx >= 0 && gx < W_) {
                v = (c < CIN_)
                    ? x  [((size_t)(b * CIN_  + c)         ) * HW_ + gy * W_ + gx]
                    : psf[((size_t)(b * CPSF_ + (c - CIN_))) * HW_ + gy * W_ + gx];
            }
            patch[c][iy / 1][ix] = v;   // flat layout matches TDM tile order
            (void)0;
        }
    }
    __syncthreads();

    const int ty = threadIdx.y, tx = threadIdx.x;
    float accO[18], accM[9];
#pragma unroll
    for (int o = 0; o < 18; ++o) accO[o] = ob[o];
#pragma unroll
    for (int o = 0; o < 9;  ++o) accM[o] = mb[o];

    for (int c = 0; c < CXT_; ++c) {
#pragma unroll
        for (int ky = 0; ky < 3; ++ky) {
#pragma unroll
            for (int kx = 0; kx < 3; ++kx) {
                float xv = patch[c][ty + ky][tx + kx];
                int   wi = (c * 3 + ky) * 3 + kx;          // within [40][3][3]
#pragma unroll
                for (int o = 0; o < 18; ++o) accO[o] = fmaf(xv, ow[o * 360 + wi], accO[o]);
#pragma unroll
                for (int o = 0; o < 9;  ++o) accM[o] = fmaf(xv, mw[o * 360 + wi], accM[o]);
            }
        }
    }

    const int hw = (ty0 + ty) * W_ + (tx0 + tx);
#pragma unroll
    for (int o = 0; o < 18; ++o) offo[((size_t)(b * 18 + o)) * HW_ + hw] = accO[o];
#pragma unroll
    for (int o = 0; o < 9;  ++o) msko[((size_t)(b * 9  + o)) * HW_ + hw] = 2.f / (1.f + __expf(-accM[o]));
}

// ---------------------------------------------------------------------------
// Kernel 2: one-time weight f32 -> bf16 (A-matrix of the GEMM, [64][288]).
// ---------------------------------------------------------------------------
__global__ __launch_bounds__(256) void wcvt_kernel(const float* __restrict__ w,
                                                   __bf16* __restrict__ wb)
{
    int i = blockIdx.x * 256 + threadIdx.x;
    if (i < COUT_ * KDIM_) wb[i] = f2bf(w[i]);
}

// ---------------------------------------------------------------------------
// Kernel 3: fused deformable gather + WMMA GEMM.
// Block handles TILE_N=64 pixels for all 64 output channels.
//   stage 1: per (tap k, pixel j): bilinear coords + weights*mask -> LDS
//   stage 2: gather x -> bf16 val tile, transposed [pixel][K] in LDS
//   stage 3: 8 waves, each: N-tile = wave&3, M-tiles = (wave>>2)*2 + {0,1};
//            K-loop of 9 x v_wmma_f32_16x16x32_bf16
// ---------------------------------------------------------------------------
__global__ __launch_bounds__(256) void deform_wmma_kernel(
    const float* __restrict__ x, const float* __restrict__ offs,
    const float* __restrict__ msk, const __bf16* __restrict__ wbf,
    float* __restrict__ out)
{
    __shared__ __bf16 vlds[TILE_N][RS_];     // 37888 B
    __shared__ int    pc[KK_ * TILE_N][2];   //  4608 B
    __shared__ float  pw[KK_ * TILE_N][4];   //  9216 B

    const int tid  = threadIdx.x;
    const int base = blockIdx.x * TILE_N;

    // ---- stage 1: sampling coordinates & weights per (k, pixel) ----
    for (int i = tid; i < KK_ * TILE_N; i += 256) {
        int k = i >> 6, j = i & (TILE_N - 1);
        int p = base + j;
        int b = p / HW_; int hw = p - b * HW_;
        int ho = hw / W_, wo = hw - ho * W_;
        float dy = offs[((size_t)(b * 18 + 2 * k    )) * HW_ + hw];
        float dx = offs[((size_t)(b * 18 + 2 * k + 1)) * HW_ + hw];
        float m  = msk [((size_t)(b * KK_ + k       )) * HW_ + hw];
        float py = (float)(ho - 1 + k / 3) + dy;
        float px = (float)(wo - 1 + k % 3) + dx;
        float fy0 = floorf(py), fx0 = floorf(px);
        float fy = py - fy0, fx = px - fx0;
        pc[i][0] = (int)fy0;
        pc[i][1] = (int)fx0;
        pw[i][0] = (1.f - fy) * (1.f - fx) * m;
        pw[i][1] = (1.f - fy) * fx * m;
        pw[i][2] = fy * (1.f - fx) * m;
        pw[i][3] = fy * fx * m;
    }
    __syncthreads();

    // ---- stage 2: bilinear gather -> transposed bf16 val tile ----
    for (int i = tid; i < KDIM_ * TILE_N; i += 256) {
        int r = i >> 6, j = i & (TILE_N - 1);   // r = c*9 + k (einsum K order)
        int c = r / KK_, k = r - c * KK_;
        int pi = (k << 6) + j;
        int p = base + j;
        int b = p / HW_;
        int y0 = pc[pi][0], x0 = pc[pi][1];
        const float* xp = x + ((size_t)(b * CIN_ + c)) * HW_;
        bool yi0 = (y0 >= 0) & (y0 < H_);
        bool yi1 = (y0 + 1 >= 0) & (y0 + 1 < H_);
        bool xi0 = (x0 >= 0) & (x0 < W_);
        bool xi1 = (x0 + 1 >= 0) & (x0 + 1 < W_);
        float v00 = (yi0 && xi0) ? xp[ y0      * W_ + x0    ] : 0.f;
        float v01 = (yi0 && xi1) ? xp[ y0      * W_ + x0 + 1] : 0.f;
        float v10 = (yi1 && xi0) ? xp[(y0 + 1) * W_ + x0    ] : 0.f;
        float v11 = (yi1 && xi1) ? xp[(y0 + 1) * W_ + x0 + 1] : 0.f;
        float val = pw[pi][0] * v00 + pw[pi][1] * v01 + pw[pi][2] * v10 + pw[pi][3] * v11;
        vlds[j][r] = f2bf(val);
    }
    __syncthreads();

    // ---- stage 3: WMMA ----
    const int wave = tid >> 5, lane = tid & 31;
    const int lrow = lane & 15, g = lane >> 4;   // ISA 16-bit A/B lane layout
    const int nt   = wave & 3;
    const int mt0  = (wave >> 2) * 2;
    const int j    = nt * 16 + lrow;

    v8f acc0 = {}, acc1 = {};
    for (int ks = 0; ks < KDIM_ / 32; ++ks) {
        v16bf af0, af1, bf;
        const __bf16* vrow = &vlds[j][ks * 32];
        const __bf16* ar0  = wbf + (size_t)(mt0 * 16 + lrow) * KDIM_ + ks * 32;
        const __bf16* ar1  = ar0 + 16 * KDIM_;
#pragma unroll
        for (int i = 0; i < 8; ++i) {
            int kk = ((i & 4) ? 16 : 0) + g * 8 + (i & 3) * 2;
            bf [2 * i] = vrow[kk]; bf [2 * i + 1] = vrow[kk + 1];
            af0[2 * i] = ar0[kk];  af0[2 * i + 1] = ar0[kk + 1];
            af1[2 * i] = ar1[kk];  af1[2 * i + 1] = ar1[kk + 1];
        }
        acc0 = __builtin_amdgcn_wmma_f32_16x16x32_bf16(false, af0, false, bf,
                                                       (short)0, acc0, false, false);
        acc1 = __builtin_amdgcn_wmma_f32_16x16x32_bf16(false, af1, false, bf,
                                                       (short)0, acc1, false, false);
    }

    // ---- store: C/D layout VGPR v -> M = v + 8*g, N = lane&15 ----
    int p = base + j;
    int b = p / HW_; int hw = p - b * HW_;
    float* obase = out + (size_t)b * COUT_ * HW_ + hw;
#pragma unroll
    for (int v = 0; v < 8; ++v) {
        obase[(size_t)(mt0 * 16      + v + g * 8) * HW_] = acc0[v];
        obase[(size_t)(mt0 * 16 + 16 + v + g * 8) * HW_] = acc1[v];
    }
}

// ---------------------------------------------------------------------------
extern "C" void kernel_launch(void* const* d_in, const int* in_sizes, int n_in,
                              void* d_out, int out_size, void* d_ws, size_t ws_size,
                              hipStream_t stream)
{
    (void)in_sizes; (void)n_in; (void)out_size; (void)ws_size;
    const float* x        = (const float*)d_in[0];
    const float* psf      = (const float*)d_in[1];
    const float* offset_w = (const float*)d_in[2];
    const float* offset_b = (const float*)d_in[3];
    const float* mod_w    = (const float*)d_in[4];
    const float* mod_b    = (const float*)d_in[5];
    const float* weight   = (const float*)d_in[6];
    float* out = (float*)d_out;

    const size_t offElems = (size_t)B_ * 18 * HW_;   // 3,612,672 f32
    const size_t mskElems = (size_t)B_ * 9  * HW_;   // 1,806,336 f32
    float*  off_buf = (float*)d_ws;
    float*  msk_buf = off_buf + offElems;
    __bf16* wbf     = (__bf16*)(msk_buf + mskElems); // 18,432 bf16

    offmask_conv_kernel<<<dim3(W_ / 16, H_ / 16, B_), dim3(16, 16, 1), 0, stream>>>(
        x, psf, offset_w, offset_b, mod_w, mod_b, off_buf, msk_buf);

    wcvt_kernel<<<(COUT_ * KDIM_ + 255) / 256, 256, 0, stream>>>(weight, wbf);

    deform_wmma_kernel<<<(B_ * HW_) / TILE_N, 256, 0, stream>>>(
        x, off_buf, msk_buf, wbf, out);
}
